// SimpleLSTM_22522808500959
// MI455X (gfx1250) — matrix operand order repaired and checked by reading, not compiled, not run on hardware
//
#include <hip/hip_runtime.h>

// ---------------------------------------------------------------------------
// 2-layer LSTM (Keras gate order i,f,g,o) + Dense(1, relu), fused scan.
//   Layer 1: z = [x_t | h] @ [W1;U1] + b1   (K = 64+128 = 192)
//   Layer 2: z = [h1_t | h] @ [W2;U2] + b2  (K = 128+128 = 256)
// One workgroup per 16 batch rows (WMMA M tile), full 512 gate columns.
// Wave w owns hidden columns w*16..w*16+15 of ALL FOUR gates, so gate math
// runs entirely on the four WMMA accumulators in registers; cell state and
// biases are registers. The recurrent half of the B operand (weights vs h)
// is hoisted into 128 VGPRs per lane ONCE -- it is invariant across all 1024
// steps -- removing ~half of the per-step LDS traffic and the dscnt stalls.
// Input panel is double-buffered with register prefetch.
// ---------------------------------------------------------------------------

typedef __attribute__((ext_vector_type(16))) _Float16 v16h;
typedef __attribute__((ext_vector_type(8)))  _Float16 v8h;
typedef __attribute__((ext_vector_type(4)))  _Float16 v4h;
typedef __attribute__((ext_vector_type(8)))  float    v8f;

#define BATCH 128
#define SEQT  1024
#define HID   128
#define GATES 512        // 4*H
#define NTHR  256        // 8 waves (wave32)

__device__ __forceinline__ float sigm_f(float x) {
  return 1.0f / (1.0f + __expf(-x));
}
__device__ __forceinline__ float tanh_f(float x) {
  float e = __expf(2.0f * x);
  return (e - 1.0f) / (e + 1.0f);
}

// L1: input = x (f32, width 64).  L2: input = hseq (f16, width 128).
template<bool L1>
__global__ void __launch_bounds__(NTHR, 1)
lstm_scan_kernel(const float*    __restrict__ xf32,   // L1 in
                 const _Float16* __restrict__ xf16,   // L2 in
                 const float*    __restrict__ W,      // [XK,512]
                 const float*    __restrict__ U,      // [128,512]
                 const float*    __restrict__ b,      // [512]
                 const float*    __restrict__ Wd,     // [128,1] (L2)
                 const float*    __restrict__ bd,     // [1]     (L2)
                 _Float16*       __restrict__ hseq_out, // (L1)
                 float*          __restrict__ out)      // (L2)
{
  constexpr int XK   = L1 ? 64 : 128;     // input-feature width
  constexpr int KTOT = XK + HID;          // 192 / 256
  constexpr int SW   = KTOT + 8;          // wT row stride (halves)
  constexpr int SA   = 2 * XK + HID + 8;  // aT row stride: 2 x-panels + h + pad
  constexpr int HOFF = 2 * XK;            // h region offset within aT row
  constexpr int NHC  = HID / 32;          // 4 recurrent K-chunks

  extern __shared__ char smem[];
  _Float16* wT = (_Float16*)smem;         // [512][SW]  transposed weights, f16
  _Float16* aT = wT + GATES * SW;         // [16][SA]   A panel: [x0|x1|h]

  const int tid  = threadIdx.x;
  const int lane = tid & 31;
  const int wave = tid >> 5;
  const int grp  = lane >> 4;             // half-wave
  const int l16  = lane & 15;
  const int row0 = blockIdx.x * 16;       // batch rows for this WG
  const int ccol = wave * 16 + l16;       // hidden column owned by this lane

  // ---- one-time: transposed f16 weights into LDS --------------------------
  for (int idx = tid; idx < GATES * KTOT; idx += NTHR) {
    const int n = idx / KTOT, k = idx % KTOT;
    const float v = (k < XK) ? W[k * GATES + n] : U[(k - XK) * GATES + n];
    wT[n * SW + k] = (_Float16)v;
  }
  // h0 = 0
  for (int idx = tid; idx < 16 * HID; idx += NTHR)
    aT[(idx >> 7) * SA + HOFF + (idx & 127)] = (_Float16)0.f;

  // ---- loop-invariant registers ------------------------------------------
  const float bi  = b[ccol];
  const float bfv = b[HID + ccol];
  const float bgv = b[2 * HID + ccol];
  const float bov = b[3 * HID + ccol];
  float creg[8];
#pragma unroll
  for (int r = 0; r < 8; ++r) creg[r] = 0.f;

  // staging: 16 threads per row, each covers XK/16 input columns
  const int sr = tid >> 4;                // 0..15 batch row (local)
  const int sc = (tid & 15) * (XK / 16);  // 4 cols (L1) / 8 cols (L2)

  float4 xr4;  v8h xr8;                   // prefetch registers
  if constexpr (L1) {
    const float4 v0 = *(const float4*)(xf32 + ((size_t)(row0 + sr) * SEQT) * XK + sc);
    v4h h4; h4[0]=(_Float16)v0.x; h4[1]=(_Float16)v0.y; h4[2]=(_Float16)v0.z; h4[3]=(_Float16)v0.w;
    *(v4h*)(aT + sr * SA + sc) = h4;      // panel 0 <- x(0)
    xr4 = *(const float4*)(xf32 + ((size_t)(row0 + sr) * SEQT + 1) * XK + sc); // x(1)
  } else {
    const v8h v0 = *(const v8h*)(xf16 + ((size_t)(row0 + sr) * SEQT) * HID + sc);
    *(v8h*)(aT + sr * SA + sc) = v0;
    xr8 = *(const v8h*)(xf16 + ((size_t)(row0 + sr) * SEQT + 1) * HID + sc);
  }
  __syncthreads();

  // ---- hoist recurrent B fragments (weights vs h) into registers ----------
  // Invariant across the whole scan: 4 gates x NHC chunks x 16 halves/lane.
  v16h bh[4][NHC];
#pragma unroll
  for (int hc = 0; hc < NHC; ++hc) {
    const int k0 = XK + hc * 32;
#pragma unroll
    for (int g = 0; g < 4; ++g) {
      const _Float16* bp = wT + (g * HID + ccol) * SW + k0 + grp * 16;
      const v8h blo = *(const v8h*)(bp);
      const v8h bhi = *(const v8h*)(bp + 8);
      bh[g][hc] = __builtin_shufflevector(blo, bhi,
          0,1,2,3,4,5,6,7,8,9,10,11,12,13,14,15);
    }
  }

  // ---- recurrent scan -----------------------------------------------------
  for (int t = 0; t < SEQT; ++t) {
    const int p = t & 1;                  // active x panel

    v8f acc[4];
#pragma unroll
    for (int g = 0; g < 4; ++g) {
      v8f z = {0.f,0.f,0.f,0.f,0.f,0.f,0.f,0.f};
      acc[g] = z;
    }

    // --- x-part: B from LDS ---
#pragma unroll
    for (int k0 = 0; k0 < XK; k0 += 32) {
      const _Float16* ap = aT + l16 * SA + p * XK + k0 + grp * 16;
      const v8h alo = *(const v8h*)(ap);
      const v8h ahi = *(const v8h*)(ap + 8);
      const v16h afrag = __builtin_shufflevector(alo, ahi,
          0,1,2,3,4,5,6,7,8,9,10,11,12,13,14,15);
#pragma unroll
      for (int g = 0; g < 4; ++g) {
        const _Float16* bp = wT + (g * HID + ccol) * SW + k0 + grp * 16;
        const v8h blo = *(const v8h*)(bp);
        const v8h bhi = *(const v8h*)(bp + 8);
        const v16h bfrag = __builtin_shufflevector(blo, bhi,
            0,1,2,3,4,5,6,7,8,9,10,11,12,13,14,15);
        acc[g] = __builtin_amdgcn_wmma_f32_16x16x32_f16(
            false, afrag, false, bfrag, (short)0, acc[g], false, false);
      }
    }
    // --- h-part: B resident in registers; only A comes from LDS ---
#pragma unroll
    for (int hc = 0; hc < NHC; ++hc) {
      const _Float16* ap = aT + l16 * SA + XK + (XK + hc * 32) - XK + grp * 8
                         + (HOFF - XK - HID) * 0;   // = aT + l16*SA + XK + hc*32 + grp*8
      const _Float16* ap2 = aT + l16 * SA + XK + hc * 32 + grp * 16;
      const v8h alo = *(const v8h*)(ap2);
      const v8h ahi = *(const v8h*)(ap2 + 8);
      const v16h afrag = __builtin_shufflevector(alo, ahi,
          0,1,2,3,4,5,6,7,8,9,10,11,12,13,14,15);
      (void)ap;
      acc[0] = __builtin_amdgcn_wmma_f32_16x16x32_f16(
          false, afrag, false, bh[0][hc], (short)0, acc[0], false, false);
#pragma unroll
      for (int g = 1; g < 4; ++g)
        acc[g] = __builtin_amdgcn_wmma_f32_16x16x32_f16(
            false, afrag, false, bh[g][hc], (short)0, acc[g],
            /*reuse_a=*/true, /*reuse_b=*/false);
    }
    __syncthreads();   // all waves done reading aT (x panel p, h)

    // stage x(t+1) from regs into panel (t+1)&1; prefetch x(t+2)
    {
      const int pn = (t + 1) & 1;
      const int tn = (t + 2 < SEQT) ? (t + 2) : (SEQT - 1);
      if constexpr (L1) {
        v4h h4; h4[0]=(_Float16)xr4.x; h4[1]=(_Float16)xr4.y; h4[2]=(_Float16)xr4.z; h4[3]=(_Float16)xr4.w;
        *(v4h*)(aT + sr * SA + pn * XK + sc) = h4;
        xr4 = *(const float4*)(xf32 + ((size_t)(row0 + sr) * SEQT + tn) * XK + sc);
      } else {
        *(v8h*)(aT + sr * SA + pn * XK + sc) = xr8;
        xr8 = *(const v8h*)(xf16 + ((size_t)(row0 + sr) * SEQT + tn) * HID + sc);
      }
    }

    // gates fully in registers: acc[g][r] is gate g, row m=r+grp*8, col ccol
#pragma unroll
    for (int r = 0; r < 8; ++r) {
      const int m = r + grp * 8;
      const float zi = acc[0][r] + bi;
      const float zf = acc[1][r] + bfv;
      const float zg = acc[2][r] + bgv;
      const float zo = acc[3][r] + bov;
      const float ig = sigm_f(zi);
      const float fg = sigm_f(zf);
      const float og = sigm_f(zo);
      const float cn = fg * creg[r] + ig * tanh_f(zg);
      const float hn = og * tanh_f(cn);
      creg[r] = cn;
      aT[m * SA + HOFF + ccol] = (_Float16)hn;
      if constexpr (L1)
        hseq_out[(size_t)(row0 + m) * SEQT * HID + (size_t)t * HID + ccol] = (_Float16)hn;
    }
    __syncthreads();   // h(t+1) + next x panel visible before next GEMM
  }

  // ---- Dense(1) + relu on final h (L2 only) -------------------------------
  if constexpr (!L1) {
    if (tid < 16) {
      float acc2 = bd[0];
      for (int k = 0; k < HID; ++k)
        acc2 += (float)aT[tid * SA + HOFF + k] * Wd[k];
      out[row0 + tid] = fmaxf(acc2, 0.0f);
    }
  }
}

// ---------------------------------------------------------------------------
extern "C" void kernel_launch(void* const* d_in, const int* in_sizes, int n_in,
                              void* d_out, int out_size, void* d_ws, size_t ws_size,
                              hipStream_t stream) {
  (void)in_sizes; (void)n_in; (void)out_size; (void)ws_size;
  const float* x  = (const float*)d_in[0];
  const float* W1 = (const float*)d_in[1];
  const float* U1 = (const float*)d_in[2];
  const float* b1 = (const float*)d_in[3];
  const float* W2 = (const float*)d_in[4];
  const float* U2 = (const float*)d_in[5];
  const float* b2 = (const float*)d_in[6];
  const float* Wd = (const float*)d_in[7];
  const float* bd = (const float*)d_in[8];

  _Float16* hseq = (_Float16*)d_ws;   // [128,1024,128] f16 = 32 MB

  // L1: SW=200, SA=264 ; L2: SW=264, SA=392  (halves)
  const size_t smem1 = (size_t)(GATES * 200 + 16 * 264) * sizeof(_Float16); // ~208 KB
  const size_t smem2 = (size_t)(GATES * 264 + 16 * 392) * sizeof(_Float16); // ~276 KB

  lstm_scan_kernel<true><<<dim3(BATCH / 16), dim3(NTHR), smem1, stream>>>(
      x, nullptr, W1, U1, b1, nullptr, nullptr, hseq, nullptr);
  lstm_scan_kernel<false><<<dim3(BATCH / 16), dim3(NTHR), smem2, stream>>>(
      nullptr, hseq, W2, U2, b2, Wd, bd, nullptr, (float*)d_out);
}
